// StackedUnidirLSTMDecoder_34325378629736
// MI455X (gfx1250) — matrix operand order repaired and checked by reading, not compiled, not run on hardware
//
#include <hip/hip_runtime.h>
#include <hip/hip_bf16.h>

// ---------------------------------------------------------------------------
// Stacked LSTM decoder, B=64 L=4 H=1024 T=128, persistent bf16-WMMA kernel.
// Weights packed once to bf16 WMMA-B-fragment layout, interleaved so all 8
// fragments of one k-step are contiguous (scalar-base + immediate-offset
// loads). Activations in bf16 A-fragment layout; cell state + epilogue f32.
// Grid: 64 WGs (one per hidden 16-slice) x 256 threads (8 waves):
//   wave = (ks, mtile): K split in halves across two wave groups (2 waves per
//   SIMD for latency hiding), partial accumulators reduced through LDS.
// 512 sequential (t,l) phases separated by an atomic grid barrier.
// ---------------------------------------------------------------------------

#define B_   64
#define L_   4
#define H_   1024
#define T_   128
#define GH   (4 * H_)       // 4096 gate rows
#define KT   (H_ / 32)      // 32 K-tiles of 32
#define KH   (KT / 2)       // 16 K-tiles per wave group
#define NWG  64             // one WG per hidden 16-tile (1024/16)
#define WGSZ 256            // 8 waves of 32

typedef __attribute__((ext_vector_type(16))) __bf16 v16bf;
typedef __attribute__((ext_vector_type(8)))  float  v8f;

// workspace layout (bytes)
#define WPACK_BYTES ((size_t)L_ * 2 * GH * H_ * 2)            // 67,108,864
#define CST_BYTES   ((size_t)L_ * B_ * H_ * 4)                //  1,048,576
#define HFRAG_BYTES ((size_t)L_ * 2 * B_ * H_ * 2)            //  1,048,576 (double buffered)
#define XFRAG_BYTES ((size_t)B_ * H_ * 2)                     //    131,072
#define OFF_W  256
#define OFF_C  (OFF_W + WPACK_BYTES)
#define OFF_HF (OFF_C + CST_BYTES)
#define OFF_XF (OFF_HF + HFRAG_BYTES)

__device__ __forceinline__ unsigned short f2bf(float f) {
    unsigned u = __float_as_uint(f);
    return (unsigned short)((u + 0x7FFFu + ((u >> 16) & 1u)) >> 16);  // RNE
}

__device__ __forceinline__ float sigmoidf_(float x) {
    return 1.0f / (1.0f + __expf(-x));
}

// A-fragment (16-bit, 16x32 MxK, ISA 7.12.2) element index for (b,h) in a
// B_xH_ matrix. Frag unit: [mtile][ktile][lane 0..31][j 0..15] = 512 bf16.
__device__ __forceinline__ unsigned afrag_elem(int b, int h) {
    int mt = b >> 4, kt = h >> 5, kl = h & 31;
    int lane = (((kl >> 3) & 1) << 4) | (b & 15);   // K in [8,16)|[24,32) -> hi lanes
    int j    = (kl & 7) | (((kl >> 4) & 1) << 3);   // second 8 elems = K+16
    return ((unsigned)(mt * KT + kt) * 32u + (unsigned)lane) * 16u + (unsigned)j;
}

// --------------------------- init: states + barrier -------------------------
__global__ void lstm_init_kernel(const float* __restrict__ x,
                                 const float* __restrict__ h0,
                                 unsigned short* __restrict__ xfrag,
                                 unsigned short* __restrict__ hfrag,
                                 unsigned* __restrict__ cnt) {
    unsigned tid = blockIdx.x * blockDim.x + threadIdx.x;
    if (tid == 0) *cnt = 0u;  // reset grid barrier every launch (graph-safe)
    if (tid < (unsigned)(L_ * B_ * H_)) {
        int l = tid / (B_ * H_);
        int r = tid % (B_ * H_);
        int b = r / H_, h = r % H_;
        // h0 -> hfrag[l][buf 0] (read at t=0); buf 1 is written before first read
        hfrag[(unsigned)(l * 2) * (B_ * H_) + afrag_elem(b, h)] = f2bf(h0[tid]);
    }
    if (tid < (unsigned)(B_ * H_)) {
        int b = tid / H_, h = tid % H_;
        xfrag[afrag_elem(b, h)] = f2bf(x[tid]);
    }
}

// ------------------- pack weights to bf16 B-fragment layout -----------------
// Interleaved layout: [l][ht 0..63][kt 0..31][q 0..7][lane 0..31][j 0..15]
// where q = gate*2 + {0:W_ih, 1:W_hh}. All 8 frags of a k-step are contiguous
// (8 KB block) so the main loop loads them with immediate offsets.
// B-frag (16-bit, 32x16 KxN): lane&15 = column, j<8: K = 8*(lane>>4)+j, j>=8: K = 16+8*(lane>>4)+(j-8).
// Flat thread id order == fragment memory order.
__global__ void lstm_pack_w_kernel(const float* __restrict__ Wih,
                                   const float* __restrict__ Whh,
                                   unsigned short* __restrict__ wp) {
    unsigned id = blockIdx.x * blockDim.x + threadIdx.x;
    if (id >= 4u * 64u * KT * 8u * 32u) return;
    unsigned lane = id & 31u;
    unsigned r = id >> 5;
    unsigned q  = r & 7u;   r >>= 3;          // gate*2 + mat
    unsigned kt = r & 31u;  r >>= 5;          // KT == 32
    unsigned ht = r & 63u;
    unsigned l  = r >> 6;
    unsigned g   = q >> 1;
    unsigned mat = q & 1u;
    unsigned n     = g * (unsigned)H_ + ht * 16u + (lane & 15u);  // gate row in 4H
    unsigned kbase = kt * 32u + ((lane >> 4) << 3);
    const float* W = (mat == 0 ? Wih : Whh) + ((size_t)(l * GH + n)) * H_ + kbase;
    unsigned short* dst = wp + (size_t)id * 16u;
#pragma unroll
    for (int j = 0; j < 16; ++j) dst[j] = f2bf(W[(j & 7) + ((j >> 3) << 4)]);  // two 8-float K runs, 16 apart
}

// ------------------------------- grid barrier -------------------------------
__device__ __forceinline__ void grid_sync(unsigned* cnt, unsigned target) {
    __syncthreads();
    __threadfence();   // every wave flushes its stores to device scope
    if (threadIdx.x == 0) {
        __hip_atomic_fetch_add(cnt, 1u, __ATOMIC_RELEASE, __HIP_MEMORY_SCOPE_AGENT);
        while (__hip_atomic_load(cnt, __ATOMIC_ACQUIRE, __HIP_MEMORY_SCOPE_AGENT) < target) {
            __builtin_amdgcn_s_sleep(2);
        }
    }
    __syncthreads();
}

// ------------------------------- main kernel --------------------------------
__global__ __launch_bounds__(WGSZ, 1) void lstm_main_kernel(
    const unsigned short* __restrict__ wp,
    unsigned short* __restrict__ hfrag,
    const unsigned short* __restrict__ xfrag,
    float* __restrict__ cst,
    const float* __restrict__ c0,
    const float* __restrict__ bih,
    const float* __restrict__ bhh,
    float* __restrict__ out,
    unsigned* __restrict__ cnt) {

    const int ht    = blockIdx.x;            // hidden tile, 0..63
    const int wave  = threadIdx.x >> 5;
    const int mtile = wave & 3;              // batch tile, 0..3
    const int ks    = wave >> 2;             // K half, 0..1
    const int lane  = threadIdx.x & 31;
    const int ncol  = ht * 16 + (lane & 15); // hidden column owned in C-layout
    const v16bf* wpv = (const v16bf*)wp;     // one v16bf = one fragment-lane (32 B)

    __shared__ v8f red[4][4][32];            // [mtile][gate][lane] partials, 16 KB

    // biases are loop-invariant over t: hoist all L*4 sums into registers
    float bsum[L_][4];
#pragma unroll
    for (int l = 0; l < L_; ++l)
#pragma unroll
        for (int g = 0; g < 4; ++g)
            bsum[l][g] = bih[l * GH + g * H_ + ncol] + bhh[l * GH + g * H_ + ncol];

    // A-frag store coordinates for column ncol (K side of next matmul)
    const int kl = ncol & 31;
    const unsigned alane_hi = (unsigned)(((kl >> 3) & 1) << 4);
    const unsigned aj       = (unsigned)((kl & 7) | (((kl >> 4) & 1) << 3));

    unsigned phase = 0;
    for (int t = 0; t < T_; ++t) {
        for (int l = 0; l < L_; ++l) {
            // layer input fragments (A matrix of x-GEMM)
            const unsigned short* xf16 =
                (l == 0) ? (t == 0 ? xfrag
                                   : hfrag + (size_t)((L_ - 1) * 2 + (t & 1)) * (B_ * H_))
                         : hfrag + (size_t)((l - 1) * 2 + ((t + 1) & 1)) * (B_ * H_);
            const unsigned short* hf16 =
                hfrag + (size_t)(l * 2 + (t & 1)) * (B_ * H_);

            // per-lane streaming pointers over this wave's K half:
            // constant stride per k-step, fragment loads use immediate offsets
            const v16bf* xrow =
                (const v16bf*)xf16 + (unsigned)(mtile * KT + ks * KH) * 32u + lane;
            const v16bf* hrow =
                (const v16bf*)hf16 + (unsigned)(mtile * KT + ks * KH) * 32u + lane;
            const v16bf* wrow =
                wpv + ((unsigned)(l * 64 + ht) * KT + (unsigned)(ks * KH)) * 8u * 32u + lane;

            v8f acc[4] = {v8f{}, v8f{}, v8f{}, v8f{}};  // i,f,g,o tiles (f32)

#pragma unroll 2
            for (int kt = 0; kt < KH; ++kt) {
                v16bf ax = xrow[0];
                v16bf ah = hrow[0];
#pragma unroll
                for (int g = 0; g < 4; ++g) {
                    v16bf bi = wrow[(g * 2 + 0) * 32];   // immediate offset g*2048
                    acc[g] = __builtin_amdgcn_wmma_f32_16x16x32_bf16(
                        false, ax, false, bi, (short)0, acc[g], false, false);
                    v16bf bh = wrow[(g * 2 + 1) * 32];   // immediate offset g*2048+1024
                    acc[g] = __builtin_amdgcn_wmma_f32_16x16x32_bf16(
                        false, ah, false, bh, (short)0, acc[g], false, false);
                }
                xrow += 32;        // +1024 B
                hrow += 32;        // +1024 B
                wrow += 8 * 32;    // +8192 B
            }

            // K-split reduction through LDS: upper-half waves publish partials
            if (ks == 1) {
#pragma unroll
                for (int g = 0; g < 4; ++g) red[mtile][g][lane] = acc[g];
            }
            __syncthreads();

            if (ks == 0) {
#pragma unroll
                for (int g = 0; g < 4; ++g) acc[g] += red[mtile][g][lane];

                // epilogue: gate nonlinearities + cell update (f32)
                unsigned short* hnew =
                    hfrag + (size_t)(l * 2 + ((t + 1) & 1)) * (B_ * H_);
#pragma unroll
                for (int v = 0; v < 8; ++v) {
                    int m = mtile * 16 + ((lane >> 4) << 3) + v;   // C-layout row
                    float gi = sigmoidf_(acc[0][v] + bsum[l][0]);
                    float gf = sigmoidf_(acc[1][v] + bsum[l][1]);
                    float gg = tanhf(acc[2][v] + bsum[l][2]);
                    float go = sigmoidf_(acc[3][v] + bsum[l][3]);
                    unsigned cidx = (unsigned)(l * B_ + m) * H_ + ncol;
                    float cold = (t == 0) ? c0[cidx] : cst[cidx];
                    float cn = gf * cold + gi * gg;
                    cst[cidx] = cn;
                    float hval = go * tanhf(cn);
                    // scatter h into A-frag layout for next consumer
                    unsigned aidx =
                        ((unsigned)((m >> 4) * KT + (ncol >> 5)) * 32u +
                         (alane_hi | (unsigned)(m & 15))) * 16u + aj;
                    hnew[aidx] = f2bf(hval);
                    if (l == L_ - 1)
                        out[((size_t)m * T_ + t) * H_ + ncol] = hval;  // (B,T,H)
                }
            }

            ++phase;
            grid_sync(cnt, phase * NWG);
        }
    }
}

// ------------------------------- launcher -----------------------------------
extern "C" void kernel_launch(void* const* d_in, const int* in_sizes, int n_in,
                              void* d_out, int out_size, void* d_ws, size_t ws_size,
                              hipStream_t stream) {
    const float* x   = (const float*)d_in[0];
    const float* h0  = (const float*)d_in[1];
    const float* c0  = (const float*)d_in[2];
    const float* Wih = (const float*)d_in[3];
    const float* Whh = (const float*)d_in[4];
    const float* bih = (const float*)d_in[5];
    const float* bhh = (const float*)d_in[6];
    float* out = (float*)d_out;

    char* ws = (char*)d_ws;
    unsigned*       cnt   = (unsigned*)ws;
    unsigned short* wp    = (unsigned short*)(ws + OFF_W);
    float*          cst   = (float*)(ws + OFF_C);
    unsigned short* hfrag = (unsigned short*)(ws + OFF_HF);
    unsigned short* xfrag = (unsigned short*)(ws + OFF_XF);

    // 1) reset barrier + pack x/h0 activations into A-frag bf16 layout
    lstm_init_kernel<<<(L_ * B_ * H_ + 255) / 256, 256, 0, stream>>>(
        x, h0, xfrag, hfrag, cnt);

    // 2) pack weights into interleaved B-frag bf16 layout (one-time HBM pass)
    lstm_pack_w_kernel<<<(4u * 64u * KT * 8u * 32u) / 256u, 256, 0, stream>>>(
        Wih, Whh, wp);

    // 3) persistent recurrence: 512 phases, atomic grid barrier between phases
    lstm_main_kernel<<<NWG, WGSZ, 0, stream>>>(
        wp, hfrag, xfrag, cst, c0, bih, bhh, out, cnt);
}
